// model_quantization_87883620811524
// MI455X (gfx1250) — matrix operand order repaired and checked by reading, not compiled, not run on hardware
//
#include <hip/hip_runtime.h>
#include <hip/hip_bf16.h>

typedef __attribute__((ext_vector_type(16))) _Float16 v16h;
typedef __attribute__((ext_vector_type(4)))  _Float16 v4h;
typedef __attribute__((ext_vector_type(8)))  float    v8f;

#define NCODE 256
#define DVEC  64
#define MT    4            // M-tiles (of 16 rows) per wave
#define SCALE     1.0e4f
#define INV2SCALE 2.0e-4f  // 2 / SCALE

// LDS: codebook in WMMA-B layout (f16, scaled) + per-codeword ||c||^2 (f32, exact)
// B layout used here, contiguous k matching A (32x16 B tile, wave32):
//   lane = n%16 + (k32>=16 ? 16 : 0), element j = k32&15  (j>>1 = VGPR, j&1 = half)
__global__ __launch_bounds__(256) void vq_wmma_kernel(
    const float* __restrict__ x,
    const float* __restrict__ codebook,
    float* __restrict__ out_q,     // [rows*64]
    float* __restrict__ out_id,    // [rows] (ids as float)
    int rows)
{
    __shared__ __align__(32) _Float16 sB[NCODE * DVEC]; // 32 KB
    __shared__ float sCsq[NCODE];                       // 1 KB

    const int tid = threadIdx.x;

    // ---- stage codebook into LDS in B-operand layout, float4 granularity ----
    // Each aligned 4-float chunk of a codeword maps to 4 consecutive f16 slots.
    {
        const float4* cb4 = (const float4*)codebook;
        #pragma unroll
        for (int i = 0; i < (NCODE * DVEC / 4) / 256; ++i) {   // 16 iters
            int f = i * 256 + tid;           // float4 index
            int n = (f * 4) >> 6;
            int k = (f * 4) & 63;
            float4 c = cb4[f];
            int t    = n >> 4;
            int h    = k >> 5;
            int k32  = k & 31;
            int lane = (n & 15) + ((k32 >= 16) ? 16 : 0);
            int j    = k32 & 15;
            v4h hv;
            hv[0] = (_Float16)(c.x * SCALE);
            hv[1] = (_Float16)(c.y * SCALE);
            hv[2] = (_Float16)(c.z * SCALE);
            hv[3] = (_Float16)(c.w * SCALE);
            *(v4h*)&sB[(((t * 2 + h) * 32 + lane) << 4) + j] = hv;
        }
    }
    // exact fp32 ||c||^2 (one codeword per thread, float4 reads)
    {
        const float4* row4 = (const float4*)(codebook + tid * DVEC);
        float s = 0.f;
        #pragma unroll
        for (int i = 0; i < DVEC / 4; ++i) {
            float4 c = row4[i];
            s += c.x * c.x + c.y * c.y + c.z * c.z + c.w * c.w;
        }
        sCsq[tid] = s;
    }
    __syncthreads();

    const int wave    = tid >> 5;
    const int lane    = tid & 31;
    const int half    = lane >> 4;          // 0: rows 0-7 of tile, 1: rows 8-15
    const int lane16  = lane & 15;
    const int rowbase = (blockIdx.x * 8 + wave) * (MT * 16);

    // ---- load A tiles (MT x 16x32 f16, two K-halves) in contiguous-k layout ----
    const int koff = half * 16;             // upper lanes hold K+16..K+31
    v16h a[MT][2];
    #pragma unroll
    for (int m = 0; m < MT; ++m) {
        const float* xr = x + (size_t)(rowbase + m * 16 + lane16) * DVEC;
        #pragma unroll
        for (int h = 0; h < 2; ++h) {
            const int base = h * 32 + koff;
            #pragma unroll
            for (int j = 0; j < 16; ++j)     // element j -> K = base + j
                a[m][h][j] = (_Float16)xr[base + j];
        }
    }

    // ---- 16 n-tiles: 1 B-tile pair feeds MT*2 WMMAs; running argmax per row ----
    float bestv[MT][8];
    int   besti[MT][8];
    #pragma unroll
    for (int m = 0; m < MT; ++m)
        #pragma unroll
        for (int r = 0; r < 8; ++r) { bestv[m][r] = -3.0e38f; besti[m][r] = 0; }

    #pragma unroll 2
    for (int t = 0; t < 16; ++t) {
        const v16h b0 = *(const v16h*)&sB[((t * 2 + 0) * 32 + lane) << 4];
        const v16h b1 = *(const v16h*)&sB[((t * 2 + 1) * 32 + lane) << 4];
        const int   n  = t * 16 + lane16;
        const float cs = sCsq[n];
        #pragma unroll
        for (int m = 0; m < MT; ++m) {
            v8f acc = {};
            acc = __builtin_amdgcn_wmma_f32_16x16x32_f16(false, a[m][0], false, b0,
                                                         (short)0, acc, false, false);
            acc = __builtin_amdgcn_wmma_f32_16x16x32_f16(false, a[m][1], false, b1,
                                                         (short)0, acc, false, false);
            #pragma unroll
            for (int r = 0; r < 8; ++r) {
                float s = INV2SCALE * acc[r] - cs;  // 2*x.c - ||c||^2 (monotone in -L2)
                if (s > bestv[m][r]) { bestv[m][r] = s; besti[m][r] = n; }
            }
        }
    }

    // ---- per tile: butterfly reduce across 16-lane halves, then write out ----
    #pragma unroll
    for (int m = 0; m < MT; ++m) {
        #pragma unroll
        for (int off = 8; off > 0; off >>= 1) {
            #pragma unroll
            for (int r = 0; r < 8; ++r) {
                float ov = __shfl_xor(bestv[m][r], off, 32);
                int   oi = __shfl_xor(besti[m][r], off, 32);
                if (ov > bestv[m][r] || (ov == bestv[m][r] && oi < besti[m][r])) {
                    bestv[m][r] = ov; besti[m][r] = oi;
                }
            }
        }
        const int rb = rowbase + m * 16;

        // indices (as float, matching harness float d_out)
        if (lane16 == 0) {
            #pragma unroll
            for (int r = 0; r < 8; ++r)
                out_id[rb + half * 8 + r] = (float)besti[m][r];
        }

        // gather + write q_hard: 2 floats per lane per row
        #pragma unroll
        for (int r = 0; r < 8; ++r) {
            int idlo = __shfl(besti[m][r], 0, 32);    // row r     (lower half's best)
            int idhi = __shfl(besti[m][r], 16, 32);   // row r + 8 (upper half's best)
            float2 vlo = ((const float2*)(codebook + (size_t)idlo * DVEC))[lane];
            float2 vhi = ((const float2*)(codebook + (size_t)idhi * DVEC))[lane];
            ((float2*)(out_q + (size_t)(rb + r)     * DVEC))[lane] = vlo;
            ((float2*)(out_q + (size_t)(rb + 8 + r) * DVEC))[lane] = vhi;
        }
    }
}

extern "C" void kernel_launch(void* const* d_in, const int* in_sizes, int n_in,
                              void* d_out, int out_size, void* d_ws, size_t ws_size,
                              hipStream_t stream) {
    const float* x        = (const float*)d_in[0];
    const float* codebook = (const float*)d_in[1];
    const int rows = in_sizes[0] / DVEC;           // 524288

    float* out_q  = (float*)d_out;                 // [rows*64]
    float* out_id = out_q + (size_t)rows * DVEC;   // [rows]

    const int rows_per_block = 8 * MT * 16;        // 8 waves * 64 rows = 512
    dim3 grid(rows / rows_per_block), block(256);
    vq_wmma_kernel<<<grid, block, 0, stream>>>(x, codebook, out_q, out_id, rows);
}
